// DifferentiableSuperpixelTokenizer_5394478924373
// MI455X (gfx1250) — matrix operand order repaired and checked by reading, not compiled, not run on hardware
//
#include <hip/hip_runtime.h>

typedef _Float16 v16h __attribute__((ext_vector_type(16)));
typedef _Float16 v8h  __attribute__((ext_vector_type(8)));
typedef float    v8f  __attribute__((ext_vector_type(8)));
typedef float    v2f  __attribute__((ext_vector_type(2)));
typedef int      v4i  __attribute__((ext_vector_type(4)));

constexpr int BATCH = 2;
constexpr int HH = 224;
constexpr int WW = 224;
constexpr int HW = HH * WW;          // 50176
constexpr int P_TOT = BATCH * HW;    // 100352
constexpr int MSEG = 196;
constexpr int EMB = 768;
constexpr int C1 = 64;
constexpr int KDIM = C1 * 9;         // 576
constexpr int KFUSE = 2 * EMB;       // 1536
constexpr int ROWS = BATCH * MSEG;   // 392

// ---- CDNA5 async global->LDS copy (ASYNCcnt) with safe fallback ----
// ROCm builtin signature (from round-2 diagnostic): (v4i __device__*, v4i __shared__*, int, int)
__device__ __forceinline__ void async_copy16(const void* g, void* l) {
#if __has_builtin(__builtin_amdgcn_global_load_async_to_lds_b128)
  __builtin_amdgcn_global_load_async_to_lds_b128(
      (__attribute__((address_space(1))) v4i*)(void*)g,
      (__attribute__((address_space(3))) v4i*)l, 0, 0);
#else
  *(v8h*)l = *(const v8h*)g;   // fallback: global_load_b128 + ds_store_b128
#endif
}
__device__ __forceinline__ void async_wait0() {
#if __has_builtin(__builtin_amdgcn_s_wait_asynccnt)
  __builtin_amdgcn_s_wait_asynccnt(0);
#endif
}

// ---------------- init accumulators (ws is poisoned each run) ----------------
__global__ void k_init(float* __restrict__ emb_sum, int* __restrict__ emb_max_bits,
                       float* __restrict__ counts, float* __restrict__ sims) {
  int t = blockIdx.x * 256 + threadIdx.x;          // 301056 threads
  emb_sum[t] = 0.f;
  emb_max_bits[t] = 0xFF800000;                    // -inf
  if (t < ROWS) { counts[t] = 0.f; sims[t] = 0.f; }
}

// ---------------- conv1 + BN + ReLU -> f16 NHWC ----------------
__global__ void k_conv1(const float* __restrict__ img, const float* __restrict__ w,
                        const float* __restrict__ bias, const float* __restrict__ g,
                        const float* __restrict__ bb, const float* __restrict__ m,
                        const float* __restrict__ v, _Float16* __restrict__ xh) {
  int t = blockIdx.x * 256 + threadIdx.x;          // p*64 + c
  int c = t & 63;
  int p = t >> 6;                                  // b*HW + y*WW + x
  int b = p / HW;
  int y = (p % HW) / WW;
  int x = p % WW;
  float acc = bias[c];
#pragma unroll
  for (int ci = 0; ci < 3; ++ci)
#pragma unroll
    for (int ky = 0; ky < 3; ++ky) {
      int ys = y + ky - 1;
      if ((unsigned)ys >= HH) continue;
#pragma unroll
      for (int kx = 0; kx < 3; ++kx) {
        int xs = x + kx - 1;
        if ((unsigned)xs >= WW) continue;
        acc += img[((b * 3 + ci) * HH + ys) * WW + xs] * w[((c * 3 + ci) * 3 + ky) * 3 + kx];
      }
    }
  float sc = g[c] * rsqrtf(v[c] + 1e-5f);
  float val = fmaxf(acc * sc + (bb[c] - m[c] * sc), 0.f);
  xh[t] = (_Float16)val;                           // NHWC: ((b*HH+y)*WW+x)*64 + c
}

// ---------------- pack conv2 weights into WMMA B-fragment order ----------------
// wp[((nt*18 + kc)*32 + lane)*16 + h]: B lane holds col n=nt*16+lane%16,
// K = (h/8)*16 + (lane/16)*8 + h%8 within 32-wide chunk kc (tap = kc/2, ic = (kc&1)*32 + K)
__global__ void k_packw(const float* __restrict__ w2, _Float16* __restrict__ wp) {
  int t = blockIdx.x * 256 + threadIdx.x;          // 442368 threads
  int h  = t & 15;
  int L  = (t >> 4) & 31;
  int kc = (t >> 9) % 18;
  int nt = t / (512 * 18);
  int oc  = nt * 16 + (L & 15);
  int kk  = ((h >> 3) << 4) + ((L >> 4) << 3) + (h & 7); // 0..31
  int ic  = ((kc & 1) << 5) + kk;
  int tap = kc >> 1;
  wp[t] = (_Float16)w2[oc * KDIM + ic * 9 + tap];  // [E][64][3][3] row-major
}

// ---------------- edge conv, gradient_map, similarity sums, segment copy ----------------
__global__ void k_edges(const float* __restrict__ img, const int* __restrict__ segs,
                        const float* __restrict__ ew, float* __restrict__ out,
                        float* __restrict__ counts, float* __restrict__ sims) {
  int p = blockIdx.x * 256 + threadIdx.x;          // 100352 threads
  int b = p / HW;
  int y = (p % HW) / WW;
  int x = p % WW;
  float e0 = 0.f, e1 = 0.f;
#pragma unroll
  for (int ky = 0; ky < 3; ++ky) {
    int ys = y + ky - 1;
    if ((unsigned)ys >= HH) continue;
#pragma unroll
    for (int kx = 0; kx < 3; ++kx) {
      int xs = x + kx - 1;
      if ((unsigned)xs >= WW) continue;
      int base = (b * 3) * HW + ys * WW + xs;
      float gray = (img[base] + img[base + HW] + img[base + 2 * HW]) * (1.f / 3.f);
      e0 += gray * ew[ky * 3 + kx];
      e1 += gray * ew[9 + ky * 3 + kx];
    }
  }
  float grad = sqrtf(e0 * e0 + e1 * e1 + 1e-8f);
  out[ROWS * EMB + p] = grad;                               // output 2: gradient_map
  float sim = fminf(fmaxf(1.f - grad, 0.f), 1.f);
  int id = b * MSEG + segs[p];
  atomicAdd(&counts[id], 1.f);
  atomicAdd(&sims[id], sim);
  out[ROWS * EMB + P_TOT + p] = (float)segs[p];             // output 3: segments
}

// ---------------- conv2 implicit GEMM (f16 WMMA) + fused atomic pooling ----------------
// Block = 8 waves, 128 pixel-rows x 64 channels. B panel staged to LDS in two
// 36KB phases via async global->LDS; all 8 waves share it (8x less global B traffic).
__global__ void k_conv2_wmma(const _Float16* __restrict__ xh, const _Float16* __restrict__ wp,
                             const int* __restrict__ segs, const float* __restrict__ cb2,
                             const float* __restrict__ g2, const float* __restrict__ b2,
                             const float* __restrict__ m2, const float* __restrict__ v2,
                             float* __restrict__ emb_sum, float* __restrict__ emb_max) {
  __shared__ __align__(32) _Float16 Bsh[18432];    // 4 N-tiles x 9 chunks x 32 lanes x 16h = 36KB
  int wave = threadIdx.x >> 5;
  int lane = threadIdx.x & 31;
  int lsel = lane >> 4;
  int lrow = lane & 15;
  int rowbase = (blockIdx.x * 8 + wave) * 16;      // pixel-row tile base
  int cbase   = blockIdx.y * 64;                   // channel base (4 N-tiles)

  int pa = rowbase + lrow;                         // this lane's A row (pixel)
  int ba = pa / HW;
  int ya = (pa % HW) / WW;
  int xa = pa % WW;

  v8f acc[4];
#pragma unroll
  for (int j = 0; j < 4; ++j)
#pragma unroll
    for (int i = 0; i < 8; ++i) acc[j][i] = 0.f;

  const char* wpb = (const char*)wp;
#pragma unroll
  for (int phase = 0; phase < 2; ++phase) {
    const int kc0 = phase * 9;
    // ---- stage B panel for chunks [kc0, kc0+9) : 2304 x 16B, 9 per thread ----
#pragma unroll
    for (int i = 0; i < 9; ++i) {
      int cpy = threadIdx.x + i * 256;             // 0..2303
      int j = cpy / 576, rem = cpy % 576;          // (nt,kc) block = 1024B, 9 per j
      async_copy16(wpb + ((blockIdx.y * 4 + j) * 18 + kc0) * 1024 + rem * 16,
                   (char*)Bsh + cpy * 16);
    }
    async_wait0();
    __syncthreads();

    // ---- 9 K-chunks of 32, tap geometry fully compile-time ----
#pragma unroll
    for (int kcl = 0; kcl < 9; ++kcl) {
      const int kc = kc0 + kcl;
      const int tap = kc >> 1, h2 = kc & 1;
      const int dy = tap / 3, dx = tap % 3;
      int ys = ya + dy * 2 - 2;                    // dilation 2, pad 2
      int xs = xa + dx * 2 - 2;
      v16h afrag;
      if ((unsigned)ys < (unsigned)HH && (unsigned)xs < (unsigned)WW) {
        const _Float16* ap = xh + (((ba * HH + ys) * WW + xs) * 64 + h2 * 32 + lsel * 8);
        v8h lo = *(const v8h*)ap;                  // K = lsel*8 .. +7
        v8h hi = *(const v8h*)(ap + 16);           // K = 16 + lsel*8 .. +7
        afrag = __builtin_shufflevector(lo, hi, 0,1,2,3,4,5,6,7,8,9,10,11,12,13,14,15);
      } else {
#pragma unroll
        for (int i = 0; i < 16; ++i) afrag[i] = (_Float16)0.f;
      }
#pragma unroll
      for (int j = 0; j < 4; ++j) {
        v16h bfrag = *(const v16h*)(Bsh + ((j * 9 + kcl) * 32 + lane) * 16);
        acc[j] = __builtin_amdgcn_wmma_f32_16x16x32_f16(
            false, afrag, false, bfrag, (short)0, acc[j], false, false);
      }
    }
    __syncthreads();                               // LDS reuse fence before next phase
  }

  // epilogue: bias + BN + ReLU, then segment-sum / segment-max via atomics.
  // relu => f >= 0, so signed-int atomicMax is a correct float max vs -inf init.
#pragma unroll
  for (int j = 0; j < 4; ++j) {
    int ch = cbase + j * 16 + lrow;
    float sc = g2[ch] * rsqrtf(v2[ch] + 1e-5f);
    float sh = b2[ch] - m2[ch] * sc;
    float bias = cb2[ch];
#pragma unroll
    for (int r = 0; r < 8; ++r) {
      int p = rowbase + r + lsel * 8;              // C/D layout: M = r + 8*lsel
      int id = (p / HW) * MSEG + segs[p];
      float f = fmaxf((acc[j][r] + bias) * sc + sh, 0.f);
      int off = id * EMB + ch;
      atomicAdd(&emb_sum[off], f);
      atomicMax((int*)&emb_max[off], __float_as_int(f));
    }
  }
}

// ---------------- build padded concat matrix A = [mean | max] ----------------
__global__ void k_fixup(const float* __restrict__ emb_sum, const float* __restrict__ emb_max,
                        const float* __restrict__ counts, float* __restrict__ Abuf) {
  int t = blockIdx.x * 256 + threadIdx.x;          // 400*1536 = 614400 threads
  int k = t % KFUSE;
  int row = t / KFUSE;
  float val = 0.f;
  if (row < ROWS) {
    float cnt = counts[row];
    if (k < EMB) val = emb_sum[row * EMB + k] / fmaxf(cnt, 1.f);
    else         val = (cnt > 0.f) ? emb_max[row * EMB + (k - EMB)] : 0.f;
  }
  Abuf[t] = val;
}

// ---------------- fusion GEMM (exact f32 WMMA, dual accumulators) ----------------
__global__ void k_fusion_wmma(const float* __restrict__ Abuf, const float* __restrict__ fw,
                              const float* __restrict__ fb, const float* __restrict__ pw,
                              const float* __restrict__ pb, const float* __restrict__ cent,
                              const float* __restrict__ counts, const float* __restrict__ sims,
                              float* __restrict__ out) {
  int wave = threadIdx.x >> 5;
  int lane = threadIdx.x & 31;
  int lsel = lane >> 4;
  int lidx = lane & 15;
  int tile = blockIdx.x * 8 + wave;                // 1200 tiles = 25 M x 48 N
  int mt = tile % 25;
  int nt = tile / 25;
  int n = nt * 16 + lidx;                          // output channel d
  int arow = mt * 16 + lidx;                       // A row for this lane (< 400)

  v8f acc0, acc1;
#pragma unroll
  for (int i = 0; i < 8; ++i) { acc0[i] = 0.f; acc1[i] = 0.f; }

  const float* ap = Abuf + arow * KFUSE + lsel * 2;
  const float* bp = fw + n * KFUSE + lsel * 2;
  for (int kb = 0; kb < KFUSE; kb += 8) {          // two independent WMMA chains
    v2f a0 = *(const v2f*)(ap + kb);
    v2f b0 = *(const v2f*)(bp + kb);
    acc0 = __builtin_amdgcn_wmma_f32_16x16x4_f32(false, a0, false, b0, (short)0, acc0, false, false);
    v2f a1 = *(const v2f*)(ap + kb + 4);
    v2f b1 = *(const v2f*)(bp + kb + 4);
    acc1 = __builtin_amdgcn_wmma_f32_16x16x4_f32(false, a1, false, b1, (short)0, acc1, false, false);
  }

  float fbias = fb[n];
  float pw0 = pw[n * 2], pw1 = pw[n * 2 + 1], pbias = pb[n];
#pragma unroll
  for (int r = 0; r < 8; ++r) {
    int row = mt * 16 + r + lsel * 8;
    if (row < ROWS) {
      float wk = sims[row] / fmaxf(counts[row], 1.f);
      float cx = cent[row * 2] * (1.f / 224.f);
      float cy = cent[row * 2 + 1] * (1.f / 224.f);
      out[row * EMB + n] = (acc0[r] + acc1[r] + fbias) * wk + pw0 * cx + pw1 * cy + pbias;
    }
  }
}

extern "C" void kernel_launch(void* const* d_in, const int* in_sizes, int n_in,
                              void* d_out, int out_size, void* d_ws, size_t ws_size,
                              hipStream_t stream) {
  (void)in_sizes; (void)n_in; (void)out_size; (void)ws_size;
  const float* img  = (const float*)d_in[0];
  const int*   segs = (const int*)d_in[1];
  const float* cent = (const float*)d_in[2];
  const float* c1w  = (const float*)d_in[3];
  const float* c1b  = (const float*)d_in[4];
  const float* g1   = (const float*)d_in[5];
  const float* b1   = (const float*)d_in[6];
  const float* m1   = (const float*)d_in[7];
  const float* v1   = (const float*)d_in[8];
  const float* c2w  = (const float*)d_in[9];
  const float* c2b  = (const float*)d_in[10];
  const float* g2   = (const float*)d_in[11];
  const float* b2   = (const float*)d_in[12];
  const float* m2   = (const float*)d_in[13];
  const float* v2   = (const float*)d_in[14];
  const float* ew   = (const float*)d_in[15];
  const float* pw   = (const float*)d_in[16];
  const float* pb   = (const float*)d_in[17];
  const float* fw   = (const float*)d_in[18];
  const float* fb   = (const float*)d_in[19];
  float* out = (float*)d_out;

  // workspace layout (bytes), all 256B-aligned; total ~18.6 MB
  char* ws = (char*)d_ws;
  _Float16* xh   = (_Float16*)(ws + 0);            // 2*224*224*64 f16 = 12,845,056
  _Float16* wpk  = (_Float16*)(ws + 12845056);     // 768*576 f16    =    884,736
  float* emb_sum = (float*)(ws + 13729792);        // 392*768 f32    =  1,204,224
  float* emb_max = (float*)(ws + 14934016);        // 392*768 f32    =  1,204,224
  float* counts  = (float*)(ws + 16138240);        // 392 f32 (pad 2048)
  float* sims    = (float*)(ws + 16140288);        // 392 f32 (pad 2048)
  float* Abuf    = (float*)(ws + 16142336);        // 400*1536 f32   =  2,457,600

  k_init<<<1176, 256, 0, stream>>>(emb_sum, (int*)emb_max, counts, sims);
  k_conv1<<<25088, 256, 0, stream>>>(img, c1w, c1b, g1, b1, m1, v1, xh);
  k_packw<<<1728, 256, 0, stream>>>(c2w, wpk);
  k_edges<<<392, 256, 0, stream>>>(img, segs, ew, out, counts, sims);
  dim3 grid2(784, 12);                             // 6272 M-tiles / 8 waves, 12 * 64ch
  k_conv2_wmma<<<grid2, 256, 0, stream>>>(xh, wpk, segs, c2b, g2, b2, m2, v2,
                                          emb_sum, emb_max);
  k_fixup<<<2400, 256, 0, stream>>>(emb_sum, emb_max, counts, Abuf);
  k_fusion_wmma<<<150, 256, 0, stream>>>(Abuf, fw, fb, pw, pb, cent, counts, sims, out);
}
